// Model_41824391529226
// MI455X (gfx1250) — matrix operand order repaired and checked by reading, not compiled, not run on hardware
//
#include <hip/hip_runtime.h>
#include <hip/hip_bf16.h>
#include <stdint.h>

// ---------------------------------------------------------------------------
// CDNA5 (gfx1250) GNN pipeline: GeneralConv x2 + mean-pool + MLP
// Edge/node GEMMs use v_wmma_f32_16x16x32_bf16 (K=64 -> 2 WMMAs per tile).
// ---------------------------------------------------------------------------

typedef __bf16 v16bf __attribute__((ext_vector_type(16)));
typedef float  v8f   __attribute__((ext_vector_type(8)));

#define NEG_SLOPE 0.2f

union ABFrag { v16bf v; uint4 q[2]; };

__device__ __forceinline__ unsigned short f2bf(float f) {
  unsigned u = __float_as_uint(f);
  u += 0x7fffu + ((u >> 16) & 1u);          // round-to-nearest-even
  return (unsigned short)(u >> 16);
}
__device__ __forceinline__ float bf2f(unsigned short h) {
  return __uint_as_float((unsigned)h << 16);
}
// monotone float <-> uint encoding for atomicMax-based segment max
__device__ __forceinline__ unsigned encf(float f) {
  unsigned u = __float_as_uint(f);
  return (u & 0x80000000u) ? ~u : (u | 0x80000000u);
}
__device__ __forceinline__ float decf(unsigned k) {
  unsigned u = (k & 0x80000000u) ? (k & 0x7fffffffu) : ~k;
  return __uint_as_float(u);
}

// ---------------------------------------------------------------------------
__global__ void embed_kernel(const int* __restrict__ x, const float* __restrict__ emb,
                             unsigned short* __restrict__ hbf, int N) {
  int i = blockIdx.x * blockDim.x + threadIdx.x;
  if (i >= N * 64) return;
  int n = i >> 6, d = i & 63;
  hbf[i] = f2bf(emb[(size_t)x[n] * 64 + d]);
}

// ---------------------------------------------------------------------------
// Edge message kernel: m = h[src]@Wm + ea@We + (bm+be); a = leaky_relu(m.att)
// Block = 256 threads (8 waves); each wave: 16 edges x C outputs via WMMA.
// ---------------------------------------------------------------------------
template <int C>
__global__ void __launch_bounds__(256) edge_msg_kernel(
    const unsigned short* __restrict__ hbf,   // [N,64] bf16
    const int* __restrict__ src,
    const float* __restrict__ ea,             // [E,8]
    const float* __restrict__ Wm,             // [64,C]
    const float* __restrict__ bm, const float* __restrict__ We,  // [8,C]
    const float* __restrict__ be, const float* __restrict__ att, // [C]
    unsigned short* __restrict__ m_out,       // [E,C] bf16
    float* __restrict__ a_out,                // [E]
    int E) {
  __shared__ unsigned short Wt[C * 80];       // Wt[c*80 + k] (transposed, padded)
  __shared__ float We_s[8 * C];
  __shared__ float att_s[C];
  __shared__ float bias_s[C];
  __shared__ unsigned short rows[128 * 80];   // gathered src features (padded)
  __shared__ float ea_s[128 * 8];

  const int tid = threadIdx.x;
  const int eblock = blockIdx.x * 128;

  for (int idx = tid; idx < 64 * C; idx += 256) {
    int k = idx / C, c = idx % C;
    Wt[c * 80 + k] = f2bf(Wm[idx]);
  }
  for (int idx = tid; idx < 8 * C; idx += 256) We_s[idx] = We[idx];
  for (int idx = tid; idx < C; idx += 256) {
    att_s[idx]  = att[idx];
    bias_s[idx] = bm[idx] + be[idx];
  }
  // gather 128 source rows, 8 x 16B chunks each
  for (int idx = tid; idx < 128 * 8; idx += 256) {
    int r = idx >> 3, ppart = idx & 7;
    int e = eblock + r; if (e > E - 1) e = E - 1;
    const uint4* gsrc = (const uint4*)(hbf + (size_t)src[e] * 64 + ppart * 8);
    *((uint4*)(rows + r * 80 + ppart * 8)) = *gsrc;
  }
  for (int idx = tid; idx < 128 * 8; idx += 256) {
    int r = idx >> 3, j = idx & 7;
    int e = eblock + r; if (e > E - 1) e = E - 1;
    ea_s[idx] = ea[(size_t)e * 8 + j];
  }
  __syncthreads();

  const int wave = tid >> 5, lane = tid & 31;
  const int half = lane >> 4, m16 = lane & 15;

  // A fragments (16x32 bf16 per K-block): two runs of 8 contiguous K per lane
  const unsigned short* myrow = rows + (wave * 16 + m16) * 80;
  ABFrag a0, a1;
  a0.q[0] = *(const uint4*)(myrow + half * 8);
  a0.q[1] = *(const uint4*)(myrow + 16 + half * 8);
  a1.q[0] = *(const uint4*)(myrow + 32 + half * 8);
  a1.q[1] = *(const uint4*)(myrow + 48 + half * 8);

  float score[8];
#pragma unroll
  for (int r = 0; r < 8; ++r) score[r] = 0.f;

  const int ewave = eblock + wave * 16;

#pragma unroll
  for (int ct = 0; ct < C / 16; ++ct) {
    // B fragments (32x16 bf16): two runs of 8 contiguous K of column (ct*16+m16)
    const unsigned short* wrow = Wt + (ct * 16 + m16) * 80;
    ABFrag b0, b1;
    b0.q[0] = *(const uint4*)(wrow + half * 8);
    b0.q[1] = *(const uint4*)(wrow + 16 + half * 8);
    b1.q[0] = *(const uint4*)(wrow + 32 + half * 8);
    b1.q[1] = *(const uint4*)(wrow + 48 + half * 8);

    v8f acc = {0.f, 0.f, 0.f, 0.f, 0.f, 0.f, 0.f, 0.f};
    acc = __builtin_amdgcn_wmma_f32_16x16x32_bf16(false, a0.v, false, b0.v,
                                                  (short)0, acc, false, false);
    acc = __builtin_amdgcn_wmma_f32_16x16x32_bf16(false, a1.v, false, b1.v,
                                                  (short)0, acc, false, false);

    const int col = ct * 16 + m16;
    const float attc = att_s[col];
    const float bc = bias_s[col];
#pragma unroll
    for (int r = 0; r < 8; ++r) {
      const int M = r + 8 * half;
      float v = acc[r] + bc;
#pragma unroll
      for (int j = 0; j < 8; ++j)
        v += ea_s[(wave * 16 + M) * 8 + j] * We_s[j * C + col];
      score[r] += v * attc;
      const int e = ewave + M;
      if (e < E) m_out[(size_t)e * C + col] = f2bf(v);
    }
  }
  // reduce attention dot across the 16 lanes holding one row's columns
#pragma unroll
  for (int r = 0; r < 8; ++r) {
#pragma unroll
    for (int off = 1; off < 16; off <<= 1)
      score[r] += __shfl_xor(score[r], off, 32);
  }
  if (m16 < 8) {
    float s = score[0];
#pragma unroll
    for (int r = 1; r < 8; ++r) s = (m16 == r) ? score[r] : s;
    s = (s > 0.f) ? s : NEG_SLOPE * s;
    const int e = ewave + m16 + 8 * half;
    if (e < E) a_out[e] = s;
  }
}

// ---------------------------------------------------------------------------
// Node skip GEMM: skip = h @ Ws + bs   (WMMA, identity row gather)
// ---------------------------------------------------------------------------
template <int C>
__global__ void __launch_bounds__(256) node_skip_kernel(
    const unsigned short* __restrict__ hbf, const float* __restrict__ Ws,
    const float* __restrict__ bs, float* __restrict__ skip, int N) {
  __shared__ unsigned short Wt[C * 80];
  __shared__ float bias_s[C];
  const int tid = threadIdx.x;
  for (int idx = tid; idx < 64 * C; idx += 256) {
    int k = idx / C, c = idx % C;
    Wt[c * 80 + k] = f2bf(Ws[idx]);
  }
  for (int idx = tid; idx < C; idx += 256) bias_s[idx] = bs[idx];
  __syncthreads();

  const int wave = tid >> 5, lane = tid & 31;
  const int half = lane >> 4, m16 = lane & 15;
  const int nbase = (blockIdx.x * 8 + wave) * 16;
  int row = nbase + m16; if (row > N - 1) row = N - 1;

  const unsigned short* grow = hbf + (size_t)row * 64;
  ABFrag a0, a1;
  a0.q[0] = *(const uint4*)(grow + half * 8);
  a0.q[1] = *(const uint4*)(grow + 16 + half * 8);
  a1.q[0] = *(const uint4*)(grow + 32 + half * 8);
  a1.q[1] = *(const uint4*)(grow + 48 + half * 8);

#pragma unroll
  for (int ct = 0; ct < C / 16; ++ct) {
    const unsigned short* wrow = Wt + (ct * 16 + m16) * 80;
    ABFrag b0, b1;
    b0.q[0] = *(const uint4*)(wrow + half * 8);
    b0.q[1] = *(const uint4*)(wrow + 16 + half * 8);
    b1.q[0] = *(const uint4*)(wrow + 32 + half * 8);
    b1.q[1] = *(const uint4*)(wrow + 48 + half * 8);

    v8f acc = {0.f, 0.f, 0.f, 0.f, 0.f, 0.f, 0.f, 0.f};
    acc = __builtin_amdgcn_wmma_f32_16x16x32_bf16(false, a0.v, false, b0.v,
                                                  (short)0, acc, false, false);
    acc = __builtin_amdgcn_wmma_f32_16x16x32_bf16(false, a1.v, false, b1.v,
                                                  (short)0, acc, false, false);
    const int col = ct * 16 + m16;
    const float bc = bias_s[col];
#pragma unroll
    for (int r = 0; r < 8; ++r) {
      const int n = nbase + r + 8 * half;
      if (n < N) skip[(size_t)n * C + col] = acc[r] + bc;
    }
  }
}

// ---------------------------------------------------------------------------
__global__ void seg_max_kernel(const float* __restrict__ a, const int* __restrict__ dst,
                               unsigned* __restrict__ amax, int E) {
  int e = blockIdx.x * blockDim.x + threadIdx.x;
  if (e >= E) return;
  atomicMax(&amax[dst[e]], encf(a[e]));
}

__global__ void seg_expsum_kernel(float* __restrict__ a, const int* __restrict__ dst,
                                  const unsigned* __restrict__ amax,
                                  float* __restrict__ asum, float* __restrict__ cnt, int E) {
  int e = blockIdx.x * blockDim.x + threadIdx.x;
  if (e >= E) return;
  int d = dst[e];
  float v = expf(a[e] - decf(amax[d]));
  a[e] = v;
  atomicAdd(&asum[d], v);
  atomicAdd(&cnt[d], 1.0f);
}

// weighted scatter-add of messages into destination accumulator
template <int C>
__global__ void scatter_kernel(const unsigned short* __restrict__ m_buf,
                               const float* __restrict__ a,
                               const int* __restrict__ dst,
                               const float* __restrict__ asum,
                               float* __restrict__ s_agg, int E) {
  const long long idx = (long long)blockIdx.x * blockDim.x + threadIdx.x;
  if (idx >= (long long)E * (C / 4)) return;
  const int e  = (int)(idx / (C / 4));
  const int c0 = (int)(idx % (C / 4)) * 4;
  const int d  = dst[e];
  const float alpha = a[e] / (asum[d] + 1e-16f);
  const ushort4 mv = *(const ushort4*)(m_buf + (size_t)e * C + c0);
  float* base = s_agg + (size_t)d * C + c0;
  atomicAdd(base + 0, bf2f(mv.x) * alpha);
  atomicAdd(base + 1, bf2f(mv.y) * alpha);
  atomicAdd(base + 2, bf2f(mv.z) * alpha);
  atomicAdd(base + 3, bf2f(mv.w) * alpha);
}

template <int C>
__global__ void combine_kernel(const float* __restrict__ s_agg, const float* __restrict__ cnt,
                               const float* __restrict__ skip,
                               float* __restrict__ hf, unsigned short* __restrict__ hbf, int N) {
  long long i = (long long)blockIdx.x * blockDim.x + threadIdx.x;
  if (i >= (long long)N * C) return;
  int n = (int)(i / C);
  float cn = cnt[n]; cn = cn > 1.f ? cn : 1.f;
  float v = s_agg[i] / cn + skip[i];
  v = v > 0.f ? v : 0.f;            // relu
  if (hf) hf[i] = v;
  hbf[i] = f2bf(v);
}

__global__ void pool_kernel(const float* __restrict__ h2, const int* __restrict__ batch,
                            float* __restrict__ ps, float* __restrict__ pc, int N) {
  long long i = (long long)blockIdx.x * blockDim.x + threadIdx.x;
  if (i >= (long long)N * 128) return;
  int n = (int)(i >> 7), c = (int)(i & 127);
  int g = batch[n];
  atomicAdd(&ps[g * 128 + c], h2[i]);
  if (c == 0) atomicAdd(&pc[g], 1.0f);
}

__global__ void __launch_bounds__(256) classifier_kernel(
    const float* __restrict__ ps, const float* __restrict__ pc,
    const float* __restrict__ Wc1, const float* __restrict__ bc1,
    const float* __restrict__ Wc2, const float* __restrict__ bc2,
    float* __restrict__ z, float* __restrict__ out, int G) {
  __shared__ float g_s[64 * 128];
  const int tid = threadIdx.x;
  const int Gc = G > 64 ? 64 : G;
  for (int idx = tid; idx < Gc * 128; idx += 256) {
    int gr = idx >> 7;
    float c = pc[gr]; c = c > 1.f ? c : 1.f;
    g_s[idx] = ps[idx] / c;
  }
  __syncthreads();
  for (int idx = tid; idx < Gc * 128; idx += 256) {
    int gr = idx >> 7, c = idx & 127;
    float acc = bc1[c];
    for (int k = 0; k < 128; ++k) acc += g_s[gr * 128 + k] * Wc1[k * 128 + c];
    z[idx] = acc > 0.f ? acc : 0.f;
  }
  __syncthreads();
  for (int idx = tid; idx < Gc * 4; idx += 256) {
    int gr = idx >> 2, o = idx & 3;
    float acc = bc2[o];
    for (int k = 0; k < 128; ++k) acc += z[gr * 128 + k] * Wc2[k * 4 + o];
    out[idx] = acc;
  }
}

// ---------------------------------------------------------------------------
extern "C" void kernel_launch(void* const* d_in, const int* in_sizes, int n_in,
                              void* d_out, int out_size, void* d_ws, size_t ws_size,
                              hipStream_t stream) {
  const int*   x         = (const int*)d_in[0];
  const int*   eidx      = (const int*)d_in[1];
  const float* edge_attr = (const float*)d_in[2];
  const int*   batchh    = (const int*)d_in[3];
  const float* emb       = (const float*)d_in[4];
  const float* Wm0 = (const float*)d_in[5],  *bm0 = (const float*)d_in[6];
  const float* We0 = (const float*)d_in[7],  *be0 = (const float*)d_in[8];
  const float* att0= (const float*)d_in[9];
  const float* Ws0 = (const float*)d_in[10], *bs0 = (const float*)d_in[11];
  const float* Wm1 = (const float*)d_in[12], *bm1 = (const float*)d_in[13];
  const float* We1 = (const float*)d_in[14], *be1 = (const float*)d_in[15];
  const float* att1= (const float*)d_in[16];
  const float* Ws1 = (const float*)d_in[17], *bs1 = (const float*)d_in[18];
  const float* Wc1 = (const float*)d_in[19], *bc1 = (const float*)d_in[20];
  const float* Wc2 = (const float*)d_in[21], *bc2 = (const float*)d_in[22];

  const int N = in_sizes[0];
  const int E = in_sizes[1] / 2;
  const int G = out_size / 4;
  const int* src  = eidx;
  const int* dstp = eidx + E;

  char* p = (char*)d_ws;
  auto carve = [&](size_t bytes) -> char* {
    char* r = p;
    p += (bytes + 255) & ~(size_t)255;
    return r;
  };
  unsigned short* h0bf   = (unsigned short*)carve((size_t)N * 64 * 2);
  unsigned short* h1bf   = (unsigned short*)carve((size_t)N * 64 * 2);
  unsigned short* h2bf   = (unsigned short*)carve((size_t)N * 128 * 2);
  float*          h2f    = (float*)carve((size_t)N * 128 * 4);
  unsigned short* m_buf  = (unsigned short*)carve((size_t)E * 128 * 2);
  float*          a_buf  = (float*)carve((size_t)E * 4);
  unsigned*       amax   = (unsigned*)carve((size_t)N * 4);
  float*          asum   = (float*)carve((size_t)N * 4);
  float*          cntb   = (float*)carve((size_t)N * 4);
  float*          s_agg  = (float*)carve((size_t)N * 128 * 4);
  float*          skipb  = (float*)carve((size_t)N * 128 * 4);
  float*          pool_s = (float*)carve((size_t)G * 128 * 4);
  float*          pool_c = (float*)carve((size_t)G * 4);

  embed_kernel<<<(N * 64 + 255) / 256, 256, 0, stream>>>(x, emb, h0bf, N);

  const unsigned eblocks   = (unsigned)((E + 127) / 128);
  const unsigned nblocks16 = (unsigned)(((N + 15) / 16 + 7) / 8);
  const unsigned ethreads  = (unsigned)((E + 255) / 256);

  // -------- layer 0 : 64 -> 64 --------
  hipMemsetAsync(amax, 0, (size_t)N * 4, stream);
  hipMemsetAsync(asum, 0, (size_t)N * 4, stream);
  hipMemsetAsync(cntb, 0, (size_t)N * 4, stream);
  hipMemsetAsync(s_agg, 0, (size_t)N * 64 * 4, stream);
  edge_msg_kernel<64><<<eblocks, 256, 0, stream>>>(h0bf, src, edge_attr, Wm0, bm0,
                                                   We0, be0, att0, m_buf, a_buf, E);
  seg_max_kernel<<<ethreads, 256, 0, stream>>>(a_buf, dstp, amax, E);
  seg_expsum_kernel<<<ethreads, 256, 0, stream>>>(a_buf, dstp, amax, asum, cntb, E);
  node_skip_kernel<64><<<nblocks16, 256, 0, stream>>>(h0bf, Ws0, bs0, skipb, N);
  scatter_kernel<64><<<(unsigned)(((long long)E * 16 + 255) / 256), 256, 0, stream>>>(
      m_buf, a_buf, dstp, asum, s_agg, E);
  combine_kernel<64><<<(unsigned)(((long long)N * 64 + 255) / 256), 256, 0, stream>>>(
      s_agg, cntb, skipb, (float*)nullptr, h1bf, N);

  // -------- layer 1 : 64 -> 128 --------
  hipMemsetAsync(amax, 0, (size_t)N * 4, stream);
  hipMemsetAsync(asum, 0, (size_t)N * 4, stream);
  hipMemsetAsync(cntb, 0, (size_t)N * 4, stream);
  hipMemsetAsync(s_agg, 0, (size_t)N * 128 * 4, stream);
  edge_msg_kernel<128><<<eblocks, 256, 0, stream>>>(h1bf, src, edge_attr, Wm1, bm1,
                                                    We1, be1, att1, m_buf, a_buf, E);
  seg_max_kernel<<<ethreads, 256, 0, stream>>>(a_buf, dstp, amax, E);
  seg_expsum_kernel<<<ethreads, 256, 0, stream>>>(a_buf, dstp, amax, asum, cntb, E);
  node_skip_kernel<128><<<nblocks16, 256, 0, stream>>>(h1bf, Ws1, bs1, skipb, N);
  scatter_kernel<128><<<(unsigned)(((long long)E * 32 + 255) / 256), 256, 0, stream>>>(
      m_buf, a_buf, dstp, asum, s_agg, E);
  combine_kernel<128><<<(unsigned)(((long long)N * 128 + 255) / 256), 256, 0, stream>>>(
      s_agg, cntb, skipb, h2f, h2bf, N);

  // -------- pooling + classifier --------
  hipMemsetAsync(pool_s, 0, (size_t)G * 128 * 4, stream);
  hipMemsetAsync(pool_c, 0, (size_t)G * 4, stream);
  pool_kernel<<<(unsigned)(((long long)N * 128 + 255) / 256), 256, 0, stream>>>(
      h2f, batchh, pool_s, pool_c, N);
  classifier_kernel<<<1, 256, 0, stream>>>(pool_s, pool_c, Wc1, bc1, Wc2, bc2,
                                           skipb /* z scratch */, (float*)d_out, G);
}
